// BigramLanguageModel_14010183319902
// MI455X (gfx1250) — matrix operand order repaired and checked by reading, not compiled, not run on hardware
//
#include <hip/hip_runtime.h>
#include <math.h>

// ---------------------------------------------------------------------------
// CDNA5 (gfx1250) wave32 WMMA GPT forward pass, round 2.
//  - All GEMMs: v_wmma_f32_16x16x32_f16, f32 accumulation.
//  - GEMM inner loop: double-buffered LDS fed by GLOBAL_LOAD_ASYNC_TO_LDS_B128
//    (ASYNCcnt), overlapping global->LDS DMA with WMMA compute.
//  - Weights converted once to f16 [N][K] (transposed) so tiles are raw byte
//    copies and lm_w (64MB f16) stays resident in the 192MB L2.
// ---------------------------------------------------------------------------

typedef __attribute__((ext_vector_type(16))) _Float16 v16h;
typedef __attribute__((ext_vector_type(8)))  float    v8f;

union FragH {
    v16h v;
    unsigned int u[8];
};

static __device__ inline v8f zero8() {
    v8f z;
#pragma unroll
    for (int i = 0; i < 8; ++i) z[i] = 0.0f;
    return z;
}

// A fragment (16x32 f16, MxK): lane half = lane>>4, row m = lane&15.
// VGPR j<4 : K = half*8 + 2j   ; VGPR j>=4 : K = 16 + half*8 + 2(j-4)
static __device__ inline void load_afrag(FragH& f, const _Float16* rowBase,
                                         int half, int uintOff) {
    const unsigned int* rp = (const unsigned int*)rowBase;
#pragma unroll
    for (int j = 0; j < 4; ++j) f.u[j] = rp[uintOff + half * 4 + j];
#pragma unroll
    for (int j = 0; j < 4; ++j) f.u[4 + j] = rp[uintOff + 8 + half * 4 + j];
}

// B fragment (32x16 f16, KxN) read from transposed LDS tile Bt[n][k]:
// VGPR j<4 : K = half*8 + 2j   ; VGPR j>=4 : K = 16 + half*8 + 2(j-4)
static __device__ inline void load_bfrag(FragH& f, const _Float16* rowBase,
                                         int half, int uintOff) {
    const unsigned int* rp = (const unsigned int*)rowBase;
#pragma unroll
    for (int j = 0; j < 8; ++j) f.u[j] = rp[uintOff + (j >> 2) * 8 + half * 4 + (j & 3)];
}

// Async 16B global->LDS copy (per lane). ldsOff is the wave-relative LDS byte
// address (= low 32 bits of the flat shared pointer, ISA 10.2 aperture rule).
static __device__ inline void async_copy_b128(unsigned int ldsOff,
                                              const void* gaddr) {
    asm volatile("global_load_async_to_lds_b128 %0, %1, off"
                 :
                 : "v"(ldsOff), "v"((unsigned long long)(uintptr_t)gaddr)
                 : "memory");
}

static __device__ inline void wait_async0() {
    asm volatile("s_wait_asynccnt 0" ::: "memory");
}

static __device__ inline unsigned int lds_off(const void* p) {
    return (unsigned int)(uintptr_t)p;
}

// ---------------------------------------------------------------------------
// Weight convert + transpose:  out_f16[N][K] = in_f32[K][N]
// QKV=true: in has layout [H, K, 64]; logical column n -> (h=n>>6, d=n&63).
// LDS-tiled 32x32 so both the read and the write are coalesced.
// ---------------------------------------------------------------------------
template <bool QKV>
__global__ __launch_bounds__(256) void convert_transpose_kernel(
    const float* __restrict__ in, _Float16* __restrict__ out, int K, int N) {
    __shared__ float tile[32][33];
    const int k0 = blockIdx.x * 32;
    const int n0 = blockIdx.y * 32;
    const int tx = threadIdx.x & 31;
    const int ty = threadIdx.x >> 5;  // 0..7
#pragma unroll
    for (int i = ty; i < 32; i += 8) {
        int k = k0 + i, n = n0 + tx;
        float v;
        if (QKV) {
            int h = n >> 6, d = n & 63;
            v = in[((size_t)h * K + k) * 64 + d];
        } else {
            v = in[(size_t)k * N + n];
        }
        tile[i][tx] = v;
    }
    __syncthreads();
#pragma unroll
    for (int i = ty; i < 32; i += 8) {
        int n = n0 + i, k = k0 + tx;
        out[(size_t)n * K + k] = (_Float16)tile[tx][i];
    }
}

// ---------------------------------------------------------------------------
// GEMM:  C[M,N] = A_f16[M,K] * Wt_f16[N,K]^T (+bias)(+res)(relu)
// Block tile 64x128, BK=32, 256 threads = 8 waves (4 M-strips x 2 N-strips).
// Double-buffered LDS, async global->LDS copies (3 x B128 per thread / tile).
// ---------------------------------------------------------------------------
#define BM 64
#define BN 128
#define BK 32

template <bool RELU>
__global__ __launch_bounds__(256) void gemm_wmma_f16_kernel(
    const _Float16* __restrict__ A, const _Float16* __restrict__ Wt,
    const float* __restrict__ bias, const float* __restrict__ res,
    float* __restrict__ outF, _Float16* __restrict__ outH,
    int M, int N, int K) {
    __shared__ alignas(16) _Float16 As[2][BM * BK];   // [m][k]
    __shared__ alignas(16) _Float16 Bt[2][BN * BK];   // [n][k]

    const int tid  = threadIdx.x;
    const int lane = tid & 31;
    const int wave = tid >> 5;     // 0..7
    const int wm   = wave & 3;     // 16-row strip
    const int wn   = wave >> 2;    // 0..1 -> 64-col strip
    const int bm0  = blockIdx.y * BM;
    const int bn0  = blockIdx.x * BN;
    const int lane15 = lane & 15;
    const int half   = lane >> 4;

    // per-thread async-copy assignments (16B each)
    const int arow = tid >> 2;                 // 0..63
    const int acol = (tid & 3) * 8;            // 0,8,16,24
    const int bn_0 = tid >> 2;                 // 0..63
    const int bn_1 = 64 + (tid >> 2);          // 64..127
    const int bcol = (tid & 3) * 8;
    const _Float16* aSrc  = A  + (size_t)(bm0 + arow) * K + acol;
    const _Float16* bSrc0 = Wt + (size_t)(bn0 + bn_0) * K + bcol;
    const _Float16* bSrc1 = Wt + (size_t)(bn0 + bn_1) * K + bcol;
    unsigned int aDst[2], bDst0[2], bDst1[2];
#pragma unroll
    for (int bf = 0; bf < 2; ++bf) {
        aDst[bf]  = lds_off(&As[bf][arow * BK + acol]);
        bDst0[bf] = lds_off(&Bt[bf][bn_0 * BK + bcol]);
        bDst1[bf] = lds_off(&Bt[bf][bn_1 * BK + bcol]);
    }
    auto issue_tile = [&](int bf, int k0) {
        async_copy_b128(aDst[bf],  aSrc  + k0);
        async_copy_b128(bDst0[bf], bSrc0 + k0);
        async_copy_b128(bDst1[bf], bSrc1 + k0);
    };

    v8f acc[4];
#pragma unroll
    for (int i = 0; i < 4; ++i) acc[i] = zero8();

    const int NT = K / BK;
    issue_tile(0, 0);
    for (int kt = 0; kt < NT; ++kt) {
        const int cur = kt & 1;
        wait_async0();          // this wave's copies for tile kt complete
        __syncthreads();        // every wave's copies visible; prev reads done
        if (kt + 1 < NT) issue_tile(cur ^ 1, (kt + 1) * BK);  // overlap w/ wmma

        FragH a;
        load_afrag(a, &As[cur][(wm * 16 + lane15) * BK], half, 0);
#pragma unroll
        for (int nt = 0; nt < 4; ++nt) {
            FragH b;
            load_bfrag(b, &Bt[cur][(wn * 64 + nt * 16 + lane15) * BK], half, 0);
            acc[nt] = __builtin_amdgcn_wmma_f32_16x16x32_f16(
                false, a.v, false, b.v, (short)0, acc[nt], false, false);
        }
    }

    // --- epilogue ---
#pragma unroll
    for (int nt = 0; nt < 4; ++nt) {
        int col = bn0 + wn * 64 + nt * 16 + lane15;
        float bv = bias ? bias[col] : 0.0f;
#pragma unroll
        for (int r = 0; r < 8; ++r) {
            int row = bm0 + wm * 16 + r + 8 * half;
            float v = acc[nt][r] + bv;
            if (res) v += res[(size_t)row * N + col];
            if (RELU) v = v > 0.0f ? v : 0.0f;
            if (outF) outF[(size_t)row * N + col] = v;
            if (outH) outH[(size_t)row * N + col] = (_Float16)v;
        }
    }
}

// ---------------------------------------------------------------------------
// Flash attention: one block per (b, h, 64-query chunk), 128 threads = 4 waves
// (wave w owns query rows w*16..w*16+15). Causal, online softmax.
// ---------------------------------------------------------------------------
__global__ __launch_bounds__(128) void attn_wmma_kernel(
    const _Float16* __restrict__ q, const _Float16* __restrict__ k,
    const _Float16* __restrict__ v, _Float16* __restrict__ o,
    int T, int E, float scale) {
    const int qb = blockIdx.x;   // query block (64 rows)
    const int h  = blockIdx.y;
    const int b  = blockIdx.z;

    __shared__ _Float16 Qs[64 * 64];   // [t][d]
    __shared__ _Float16 Ks[64 * 64];   // [s][d]  (== Bt layout for S=Q*K^T)
    __shared__ _Float16 Vt[64 * 64];   // [d][s]  (Bt layout for O=P*V)
    __shared__ _Float16 Ps[64 * 64];   // [t][s]
    __shared__ float    Ss[64 * 64];
    __shared__ float    mrow[64], lrow[64], resc[64];

    const int tid = threadIdx.x;
    const int lane = tid & 31;
    const int wave = tid >> 5;     // 0..3
    const int lane15 = lane & 15;
    const int half   = lane >> 4;
    const size_t headOff = (size_t)h * 64;

    for (int c = tid; c < 512; c += 128) {
        int r = c >> 3, col = (c & 7) * 8;
        *(uint4*)&Qs[r * 64 + col] =
            *(const uint4*)&q[(size_t)(b * T + qb * 64 + r) * E + headOff + col];
    }
    if (tid < 64) { mrow[tid] = -1e30f; lrow[tid] = 0.0f; }

    v8f oacc[4];
#pragma unroll
    for (int i = 0; i < 4; ++i) oacc[i] = zero8();
    __syncthreads();

    for (int jb = 0; jb <= qb; ++jb) {
        for (int c = tid; c < 512; c += 128) {
            int r = c >> 3, col = (c & 7) * 8;
            *(uint4*)&Ks[r * 64 + col] =
                *(const uint4*)&k[(size_t)(b * T + jb * 64 + r) * E + headOff + col];
        }
        for (int c = tid; c < 4096; c += 128) {
            int s = c >> 6, d = c & 63;
            Vt[d * 64 + s] = v[(size_t)(b * T + jb * 64 + s) * E + headOff + d];
        }
        __syncthreads();

        // S = Q * K^T  (K-dim 64 = 2 x K32)
        v8f sacc[4];
#pragma unroll
        for (int i = 0; i < 4; ++i) sacc[i] = zero8();
#pragma unroll
        for (int kk = 0; kk < 2; ++kk) {
            FragH a;
            load_afrag(a, &Qs[(wave * 16 + lane15) * 64], half, kk * 16);
#pragma unroll
            for (int nt = 0; nt < 4; ++nt) {
                FragH bfr;
                load_bfrag(bfr, &Ks[(nt * 16 + lane15) * 64], half, kk * 16);
                sacc[nt] = __builtin_amdgcn_wmma_f32_16x16x32_f16(
                    false, a.v, false, bfr.v, (short)0, sacc[nt], false, false);
            }
        }
#pragma unroll
        for (int nt = 0; nt < 4; ++nt) {
#pragma unroll
            for (int r = 0; r < 8; ++r) {
                int rl = wave * 16 + r + 8 * half;
                int cl = nt * 16 + lane15;
                int tq = qb * 64 + rl;
                int sg = jb * 64 + cl;
                float val = sacc[nt][r] * scale;
                if (sg > tq) val = -1e30f;
                Ss[rl * 64 + cl] = val;
            }
        }
        __syncthreads();

        if (tid < 64) {
            int row = tid;
            float mx = -1e30f;
            for (int c = 0; c < 64; ++c) mx = fmaxf(mx, Ss[row * 64 + c]);
            float mnew = fmaxf(mrow[row], mx);
            float rs = __expf(mrow[row] - mnew);
            float sum = 0.0f;
            for (int c = 0; c < 64; ++c) {
                float p = __expf(Ss[row * 64 + c] - mnew);
                Ps[row * 64 + c] = (_Float16)p;
                sum += p;
            }
            lrow[row] = lrow[row] * rs + sum;
            mrow[row] = mnew;
            resc[row] = rs;
        }
        __syncthreads();

#pragma unroll
        for (int nt = 0; nt < 4; ++nt)
#pragma unroll
            for (int r = 0; r < 8; ++r)
                oacc[nt][r] *= resc[wave * 16 + r + 8 * half];
#pragma unroll
        for (int kk = 0; kk < 2; ++kk) {
            FragH a;
            load_afrag(a, &Ps[(wave * 16 + lane15) * 64], half, kk * 16);
#pragma unroll
            for (int nt = 0; nt < 4; ++nt) {
                FragH bfr;
                load_bfrag(bfr, &Vt[(nt * 16 + lane15) * 64], half, kk * 16);
                oacc[nt] = __builtin_amdgcn_wmma_f32_16x16x32_f16(
                    false, a.v, false, bfr.v, (short)0, oacc[nt], false, false);
            }
        }
        __syncthreads();
    }

#pragma unroll
    for (int nt = 0; nt < 4; ++nt) {
#pragma unroll
        for (int r = 0; r < 8; ++r) {
            int rl = wave * 16 + r + 8 * half;
            int d  = nt * 16 + lane15;
            int t  = qb * 64 + rl;
            float val = oacc[nt][r] / lrow[rl];
            o[(size_t)(b * T + t) * E + headOff + d] = (_Float16)val;
        }
    }
}

// ---------------------------------------------------------------------------
// Support kernels
// ---------------------------------------------------------------------------
__global__ void embed_kernel(const int* __restrict__ idx,
                             const float* __restrict__ emb,
                             float* __restrict__ x, int E) {
    int row = blockIdx.x;
    const float* src = emb + (size_t)idx[row] * E;
    float* dst = x + (size_t)row * E;
    for (int c = threadIdx.x; c < E; c += blockDim.x) dst[c] = src[c];
}

__global__ __launch_bounds__(256) void layernorm_kernel(
    const float* __restrict__ x, const float* __restrict__ g,
    const float* __restrict__ bta, _Float16* __restrict__ y, int E) {
    int row = blockIdx.x;
    int tid = threadIdx.x;
    const float* xr = x + (size_t)row * E;
    __shared__ float red[256];
    float s = 0.0f;
    for (int c = tid; c < E; c += 256) s += xr[c];
    red[tid] = s;
    __syncthreads();
    for (int off = 128; off > 0; off >>= 1) {
        if (tid < off) red[tid] += red[tid + off];
        __syncthreads();
    }
    float mean = red[0] / (float)E;
    __syncthreads();
    float vs = 0.0f;
    for (int c = tid; c < E; c += 256) {
        float d = xr[c] - mean;
        vs += d * d;
    }
    red[tid] = vs;
    __syncthreads();
    for (int off = 128; off > 0; off >>= 1) {
        if (tid < off) red[tid] += red[tid + off];
        __syncthreads();
    }
    float inv = rsqrtf(red[0] / (float)E + 1e-5f);
    for (int c = tid; c < E; c += 256)
        y[(size_t)row * E + c] = (_Float16)((xr[c] - mean) * inv * g[c] + bta[c]);
}

__global__ void rope_kernel(const float* __restrict__ in,
                            _Float16* __restrict__ out, int T, int E,
                            int pairs) {
    int p = blockIdx.x * blockDim.x + threadIdx.x;
    if (p >= pairs) return;
    int ppr = E / 2;                 // pairs per row
    int row = p / ppr;               // b*T + t
    int pr  = p % ppr;               // h*32 + i
    int t = row % T;
    int i = pr & 31;                 // pair index within head (D/2 = 32)
    float ang = (float)t * __powf(10000.0f, -(float)(2 * i) / 64.0f);
    float c = __cosf(ang), s = __sinf(ang);
    size_t base = (size_t)row * E + pr * 2;
    float x0 = in[base], x1 = in[base + 1];
    out[base]     = (_Float16)(x0 * c - x1 * s);
    out[base + 1] = (_Float16)(x0 * s + x1 * c);
}

__global__ void f32_to_f16_kernel(const float* __restrict__ in,
                                  _Float16* __restrict__ out, int n) {
    int i = blockIdx.x * blockDim.x + threadIdx.x;
    if (i < n) out[i] = (_Float16)in[i];
}

__global__ void zero_loss_kernel(float* loss) { *loss = 0.0f; }

__global__ __launch_bounds__(256) void loss_kernel(
    const float* __restrict__ logits, const int* __restrict__ targets,
    float* __restrict__ loss, int V, int rows) {
    int row = blockIdx.x;
    int tid = threadIdx.x;
    const float* lr = logits + (size_t)row * V;
    __shared__ float red[256];
    float m = -1e30f;
    for (int c = tid; c < V; c += 256) m = fmaxf(m, lr[c]);
    red[tid] = m;
    __syncthreads();
    for (int off = 128; off > 0; off >>= 1) {
        if (tid < off) red[tid] = fmaxf(red[tid], red[tid + off]);
        __syncthreads();
    }
    m = red[0];
    __syncthreads();
    float s = 0.0f;
    for (int c = tid; c < V; c += 256) s += __expf(lr[c] - m);
    red[tid] = s;
    __syncthreads();
    for (int off = 128; off > 0; off >>= 1) {
        if (tid < off) red[tid] += red[tid + off];
        __syncthreads();
    }
    if (tid == 0) {
        float nll = -(lr[targets[row]] - m - __logf(red[0]));
        atomicAdd(loss, nll / (float)rows);
    }
}

// ---------------------------------------------------------------------------
// Host driver
// ---------------------------------------------------------------------------
extern "C" void kernel_launch(void* const* d_in, const int* in_sizes, int n_in,
                              void* d_out, int out_size, void* d_ws,
                              size_t ws_size, hipStream_t stream) {
    (void)in_sizes; (void)n_in; (void)out_size; (void)ws_size;

    constexpr int B = 2, T = 1024, E = 1024, H = 16, D = 64;
    constexpr int F = 4096, V = 32000, L = 6;
    constexpr int M = B * T;  // 2048

    const int*   idx    = (const int*)d_in[0];
    const int*   tgt    = (const int*)d_in[1];
    const float* emb    = (const float*)d_in[2];
    const float* wq     = (const float*)d_in[3];
    const float* wk     = (const float*)d_in[4];
    const float* wv     = (const float*)d_in[5];
    const float* wproj  = (const float*)d_in[6];
    const float* bproj  = (const float*)d_in[7];
    const float* w1     = (const float*)d_in[8];
    const float* b1     = (const float*)d_in[9];
    const float* w2     = (const float*)d_in[10];
    const float* b2     = (const float*)d_in[11];
    const float* ln1g   = (const float*)d_in[12];
    const float* ln1b   = (const float*)d_in[13];
    const float* ln2g   = (const float*)d_in[14];
    const float* ln2b   = (const float*)d_in[15];
    const float* lm_w   = (const float*)d_in[16];
    const float* lm_b   = (const float*)d_in[17];

    float* logits = (float*)d_out;
    float* loss   = logits + (size_t)M * V;

    char* ws = (char*)d_ws;
    size_t off = 0;
    auto carve = [&](size_t bytes) {
        void* p = ws + off;
        off += (bytes + 255) & ~(size_t)255;
        return p;
    };
    // activations
    float*     x   = (float*)carve((size_t)M * E * 4);
    _Float16*  y   = (_Float16*)carve((size_t)M * E * 2);
    float*     qf  = (float*)carve((size_t)M * E * 4);
    float*     kf  = (float*)carve((size_t)M * E * 4);
    _Float16*  qb  = (_Float16*)carve((size_t)M * E * 2);
    _Float16*  kb  = (_Float16*)carve((size_t)M * E * 2);
    _Float16*  vb  = (_Float16*)carve((size_t)M * E * 2);
    _Float16*  ob  = (_Float16*)carve((size_t)M * E * 2);
    _Float16*  h1  = (_Float16*)carve((size_t)M * F * 2);
    _Float16*  xb  = (_Float16*)carve((size_t)M * E * 2);
    // f16 transposed weights (per-layer buffers reused in stream order)
    _Float16*  wqT = (_Float16*)carve((size_t)E * E * 2);
    _Float16*  wkT = (_Float16*)carve((size_t)E * E * 2);
    _Float16*  wvT = (_Float16*)carve((size_t)E * E * 2);
    _Float16*  wpT = (_Float16*)carve((size_t)E * E * 2);
    _Float16*  w1T = (_Float16*)carve((size_t)E * F * 2);
    _Float16*  w2T = (_Float16*)carve((size_t)E * F * 2);
    _Float16*  lmT = (_Float16*)carve((size_t)V * E * 2);

    embed_kernel<<<M, 256, 0, stream>>>(idx, emb, x, E);

    // lm_w: convert once, stays L2-resident for the final GEMM sweep
    convert_transpose_kernel<false><<<dim3(E / 32, V / 32), 256, 0, stream>>>(
        lm_w, lmT, E, V);

    const dim3 gE(E / BN, M / BM);       // N=1024 GEMMs
    const dim3 gF(F / BN, M / BM);       // N=4096 GEMM
    const dim3 gA(T / 64, H, B);         // attention
    const int  pairs = M * E / 2;
    const float scale = 0.03125f;        // N_EMBD^-0.5 (reference scales by E)

    for (int l = 0; l < L; ++l) {
        // per-layer weight conversion (f32 -> f16, transposed to [N][K])
        convert_transpose_kernel<true><<<dim3(E / 32, E / 32), 256, 0, stream>>>(
            wq + (size_t)l * H * E * D, wqT, E, E);
        convert_transpose_kernel<true><<<dim3(E / 32, E / 32), 256, 0, stream>>>(
            wk + (size_t)l * H * E * D, wkT, E, E);
        convert_transpose_kernel<true><<<dim3(E / 32, E / 32), 256, 0, stream>>>(
            wv + (size_t)l * H * E * D, wvT, E, E);
        convert_transpose_kernel<false><<<dim3(E / 32, E / 32), 256, 0, stream>>>(
            wproj + (size_t)l * E * E, wpT, E, E);
        convert_transpose_kernel<false><<<dim3(E / 32, F / 32), 256, 0, stream>>>(
            w1 + (size_t)l * E * F, w1T, E, F);
        convert_transpose_kernel<false><<<dim3(F / 32, E / 32), 256, 0, stream>>>(
            w2 + (size_t)l * F * E, w2T, F, E);

        layernorm_kernel<<<M, 256, 0, stream>>>(x, ln1g + l * E, ln1b + l * E, y, E);

        gemm_wmma_f16_kernel<false><<<gE, 256, 0, stream>>>(
            y, wqT, nullptr, nullptr, qf, nullptr, M, E, E);
        gemm_wmma_f16_kernel<false><<<gE, 256, 0, stream>>>(
            y, wkT, nullptr, nullptr, kf, nullptr, M, E, E);
        gemm_wmma_f16_kernel<false><<<gE, 256, 0, stream>>>(
            y, wvT, nullptr, nullptr, nullptr, vb, M, E, E);

        rope_kernel<<<(pairs + 255) / 256, 256, 0, stream>>>(qf, qb, T, E, pairs);
        rope_kernel<<<(pairs + 255) / 256, 256, 0, stream>>>(kf, kb, T, E, pairs);

        attn_wmma_kernel<<<gA, 128, 0, stream>>>(qb, kb, vb, ob, T, E, scale);

        gemm_wmma_f16_kernel<false><<<gE, 256, 0, stream>>>(
            ob, wpT, bproj + l * E, x, x, nullptr, M, E, E);

        layernorm_kernel<<<M, 256, 0, stream>>>(x, ln2g + l * E, ln2b + l * E, y, E);

        gemm_wmma_f16_kernel<true><<<gF, 256, 0, stream>>>(
            y, w1T, b1 + (size_t)l * F, nullptr, nullptr, h1, M, F, E);
        gemm_wmma_f16_kernel<false><<<gE, 256, 0, stream>>>(
            h1, w2T, b2 + l * E, x, x, nullptr, M, E, F);
    }

    f32_to_f16_kernel<<<(M * E + 255) / 256, 256, 0, stream>>>(x, xb, M * E);

    const dim3 gV(V / BN, M / BM);
    gemm_wmma_f16_kernel<false><<<gV, 256, 0, stream>>>(
        xb, lmT, lm_b, nullptr, logits, nullptr, M, V, E);

    zero_loss_kernel<<<1, 1, 0, stream>>>(loss);
    loss_kernel<<<M, 256, 0, stream>>>(logits, tgt, loss, V, M);
}
